// CausalSelfAttention_541165879532
// MI455X (gfx1250) — matrix operand order repaired and checked by reading, not compiled, not run on hardware
//
#include <hip/hip_runtime.h>
#include <hip/hip_bf16.h>
#include <stdint.h>

// CDNA5 / gfx1250: wave32, WMMA 16x16x32 bf16 -> f32 accum.
typedef __attribute__((ext_vector_type(16))) __bf16 v16bf;
typedef __attribute__((ext_vector_type(8)))  float  v8f;
typedef __attribute__((ext_vector_type(4)))  unsigned int u32x4;
typedef __attribute__((ext_vector_type(8)))  int i32x8;
typedef __attribute__((ext_vector_type(4)))  int i32x4;

#define D_MODEL 1024
#define SEQ     4096
#define NHEAD   16
#define DKH     64

#if defined(__gfx1250__) && __has_builtin(__builtin_amdgcn_tensor_load_to_lds)
#define USE_TDM 1
#else
#define USE_TDM 0
#endif

union FragBF { v16bf v; uint4 q[2]; };

__device__ __forceinline__ unsigned short f2bf_bits(float f) {
    union { __bf16 b; unsigned short s; } u;
    u.b = (__bf16)f;                       // native RNE convert on gfx1250
    return u.s;
}

// ---------------------------------------------------------------------------
// GEMM: Out[M,N] = A[M,K] @ W[N,K]^T + bias[N]
// A is f32 (converted to bf16 on the fly) or already bf16; Out is bf16 or f32.
// Block tile 64(M) x 128(N), 256 threads = 8 waves, each wave a 32x32 tile.
// ---------------------------------------------------------------------------
template<bool A_BF16, bool OUT_BF16>
__global__ __launch_bounds__(256)
void gemm_bias_kernel(const void* __restrict__ Ap,
                      const float* __restrict__ W,
                      const float* __restrict__ bias,
                      void* __restrict__ Outp,
                      int M, int N, int K)
{
    __shared__ unsigned short As[64][32];    // As[m][k]   (bf16 bits)
    __shared__ unsigned short Bs[128][32];   // Bs[n][k] = W[n][k]

    const int tid  = threadIdx.x;
    const int lane = tid & 31;
    const int wave = tid >> 5;
    const int hi   = lane >> 4;       // lane half: 0 or 1
    const int col  = lane & 15;
    const int n0   = blockIdx.x * 128;
    const int m0   = blockIdx.y * 64;
    const int wm   = (wave & 1) * 32;
    const int wn   = (wave >> 1) * 32;

    v8f acc[2][2] = {};

    for (int k0 = 0; k0 < K; k0 += 32) {
        // ---- stage A tile (64x32) ----
        if (A_BF16) {
            const unsigned short* A16 = (const unsigned short*)Ap;
            int r = tid >> 2, c8 = tid & 3;                       // 256 uint4
            uint4 d = *(const uint4*)(A16 + (size_t)(m0 + r) * K + k0 + c8 * 8);
            *(uint4*)(&As[r][c8 * 8]) = d;
        } else {
            const float* A32 = (const float*)Ap;
            #pragma unroll
            for (int it = 0; it < 2; ++it) {
                int id = tid + it * 256;                          // 512 float4
                int r = id >> 3, c4 = id & 7;
                float4 d = *(const float4*)(A32 + (size_t)(m0 + r) * K + k0 + c4 * 4);
                uint2 p;
                p.x = (uint32_t)f2bf_bits(d.x) | ((uint32_t)f2bf_bits(d.y) << 16);
                p.y = (uint32_t)f2bf_bits(d.z) | ((uint32_t)f2bf_bits(d.w) << 16);
                *(uint2*)(&As[r][c4 * 4]) = p;
            }
        }
        // ---- stage B tile (128x32): Bs[n][k] = W[(n0+n)][k0+k] ----
        #pragma unroll
        for (int it = 0; it < 4; ++it) {
            int id = tid + it * 256;                              // 1024 float4
            int r = id >> 3, c4 = id & 7;
            float4 d = *(const float4*)(W + (size_t)(n0 + r) * K + k0 + c4 * 4);
            uint2 p;
            p.x = (uint32_t)f2bf_bits(d.x) | ((uint32_t)f2bf_bits(d.y) << 16);
            p.y = (uint32_t)f2bf_bits(d.z) | ((uint32_t)f2bf_bits(d.w) << 16);
            *(uint2*)(&Bs[r][c4 * 4]) = p;
        }
        __syncthreads();

        // ---- fragments per ISA layout ----
        FragBF a[2], b[2];
        #pragma unroll
        for (int i = 0; i < 2; ++i) {
            const unsigned short* pr = &As[wm + i * 16 + col][0];
            a[i].q[0] = *(const uint4*)(pr + hi * 8);            // K 0-7 / 8-15
            a[i].q[1] = *(const uint4*)(pr + 16 + hi * 8);       // K 16-23 / 24-31
        }
        #pragma unroll
        for (int j = 0; j < 2; ++j) {
            const unsigned short* pr = &Bs[wn + j * 16 + col][0];
            b[j].q[0] = *(const uint4*)(pr + hi * 8);            // K 0-7 / 8-15
            b[j].q[1] = *(const uint4*)(pr + 16 + hi * 8);       // K 16-23 / 24-31
        }
        #pragma unroll
        for (int i = 0; i < 2; ++i)
            #pragma unroll
            for (int j = 0; j < 2; ++j)
                acc[i][j] = __builtin_amdgcn_wmma_f32_16x16x32_bf16(
                    false, a[i].v, false, b[j].v, (short)0, acc[i][j], false, false);
        __syncthreads();
    }

    // ---- epilogue: D layout row = 8*hi+v, col = lane&15 ----
    #pragma unroll
    for (int i = 0; i < 2; ++i) {
        #pragma unroll
        for (int j = 0; j < 2; ++j) {
            int cg = n0 + wn + j * 16 + col;
            float bv = bias[cg];
            #pragma unroll
            for (int v = 0; v < 8; ++v) {
                int rg = m0 + wm + i * 16 + hi * 8 + v;
                float val = acc[i][j][v] + bv;
                if (OUT_BF16)
                    ((unsigned short*)Outp)[(size_t)rg * N + cg] = f2bf_bits(val);
                else
                    ((float*)Outp)[(size_t)rg * N + cg] = val;
            }
        }
    }
}

// ---------------------------------------------------------------------------
// Causal flash attention, transposed form: per key block we compute
//   S^T = K * Q^T   (rows = keys, cols = queries)
// so each lane owns one query column; softmax stats are per-lane scalars
// with a single shfl_xor(16) across the half-wave boundary.
// Output accumulates O^T = V^T * P^T (rows = head dims, cols = queries).
// K tiles are staged by the Tensor Data Mover (TENSOR_LOAD_TO_LDS).
// 4 waves/block, each wave owns a 16-query tile of a 64-query chunk.
// ---------------------------------------------------------------------------
__global__ __launch_bounds__(128)
void attn_kernel(const unsigned short* __restrict__ Q,
                 const unsigned short* __restrict__ Kg,
                 const unsigned short* __restrict__ Vg,
                 unsigned short* __restrict__ Og)
{
    __shared__ unsigned short Ks[32][64];     // Ks[key][d]
    __shared__ unsigned short Vt[64][32];     // Vt[d][key] (transposed)
    __shared__ unsigned short Ps[4][16][32];  // per-wave P^T bounce: [query][key]

    const int tid  = threadIdx.x;
    const int lane = tid & 31;
    const int wave = tid >> 5;
    const int hi   = lane >> 4;
    const int col  = lane & 15;               // this lane's query (within tile)
    const int h    = blockIdx.y;
    const int qc   = blockIdx.x;
    const int q0   = qc * 64 + wave * 16;
    const int rg   = q0 + col;                // global query row for this lane
    const size_t hoff = (size_t)h * DKH;

    // Q^T as B-fragments (held in registers whole loop):
    // b[e]: query = col, dim = kk*32 + hi*8 + e (e<8), kk*32 + 16 + hi*8 + (e-8) (e>=8).
    FragBF qb[2];
    {
        const unsigned short* qrow = Q + (size_t)rg * D_MODEL + hoff;
        #pragma unroll
        for (int kk = 0; kk < 2; ++kk) {
            qb[kk].q[0] = *(const uint4*)(qrow + kk * 32 + hi * 8);
            qb[kk].q[1] = *(const uint4*)(qrow + kk * 32 + 16 + hi * 8);
        }
    }

    v8f acc[4] = {};                          // O^T tiles: dim chunks of 16
    float m = -3.0e38f, l = 0.0f;             // per-lane (per-query) stats

    const int nkb = 2 * qc + 2;               // key blocks covering causal range
    for (int kblk = 0; kblk < nkb; ++kblk) {
        const int k0 = kblk * 32;

#if USE_TDM
        // ---- stage K tile via Tensor Data Mover: 32x64 bf16 tile ----
        if (wave == 0) {
            uint64_t ga = (uint64_t)(uintptr_t)(Kg) +
                          2ull * ((uint64_t)k0 * D_MODEL + hoff);
            uint32_t lds = (uint32_t)(uintptr_t)(&Ks[0][0]);
            u32x4 g0;
            g0[0] = 1u;                                        // count=1
            g0[1] = lds;                                       // lds_addr
            g0[2] = (uint32_t)ga;                              // global_addr[31:0]
            g0[3] = ((uint32_t)(ga >> 32) & 0x01FFFFFFu) |
                    (2u << 30);                                // addr[56:32] | type=2
            i32x8 g1;
            g1[0] = 0x00010000;          // data_size=1 (2 bytes)
            g1[1] = (int)(((unsigned)D_MODEL & 0xffffu) << 16);   // tensor_dim0 lo16
            g1[2] = (int)((((unsigned)D_MODEL >> 16) & 0xffffu) |
                          (((unsigned)SEQ & 0xffffu) << 16));     // dim0 hi16 | dim1 lo16
            g1[3] = (int)((((unsigned)SEQ >> 16) & 0xffffu) |
                          (64u << 16));                           // dim1 hi16 | tile_dim0=64
            g1[4] = 32;                  // tile_dim1=32, tile_dim2=0
            g1[5] = D_MODEL;             // tensor_dim0_stride low32
            g1[6] = 0;                   // stride0 hi16 | stride1 lo16
            g1[7] = 0;                   // stride1 hi32
            i32x4 gz4 = {0, 0, 0, 0};
            i32x8 gz8 = {0, 0, 0, 0, 0, 0, 0, 0};
            __builtin_amdgcn_tensor_load_to_lds(g0, g1, gz4, gz4, gz8, 0);
            __builtin_amdgcn_s_wait_tensorcnt((short)0);
        }
#else
        // ---- stage K tile cooperatively: Ks[key][d] ----
        #pragma unroll
        for (int it = 0; it < 2; ++it) {
            int id = tid + it * 128;          // 256 uint4
            int r = id >> 3, c8 = id & 7;
            uint4 d = *(const uint4*)(Kg + (size_t)(k0 + r) * D_MODEL + hoff + c8 * 8);
            *(uint4*)(&Ks[r][c8 * 8]) = d;
        }
#endif
        // stage V transposed: Vt[d][key]
        #pragma unroll
        for (int it = 0; it < 8; ++it) {
            int p = tid + it * 128;           // 1024 bf16 pairs
            int key = p & 31, dp = p >> 5;
            uint32_t w = *(const uint32_t*)(Vg + (size_t)(k0 + key) * D_MODEL + hoff + dp * 2);
            Vt[dp * 2][key]     = (unsigned short)(w & 0xffffu);
            Vt[dp * 2 + 1][key] = (unsigned short)(w >> 16);
        }
        __syncthreads();

        // S^T = K Q^T : two 16x16 tiles (tile t = keys t*16..t*16+15)
        // A = K rows (row = key = lane&15), accumulate over dim halves kk.
        v8f st[2] = {};
        #pragma unroll
        for (int t = 0; t < 2; ++t) {
            #pragma unroll
            for (int kk = 0; kk < 2; ++kk) {
                FragBF a;
                const unsigned short* pr = &Ks[t * 16 + col][kk * 32];
                a.q[0] = *(const uint4*)(pr + hi * 8);
                a.q[1] = *(const uint4*)(pr + 16 + hi * 8);
                st[t] = __builtin_amdgcn_wmma_f32_16x16x32_bf16(
                    false, a.v, false, qb[kk].v, (short)0, st[t], false, false);
            }
        }

        // ---- per-lane softmax over this lane's 16 key-values ----
        float s[2][8];
        float mx = -3.0e38f;
        #pragma unroll
        for (int t = 0; t < 2; ++t)
            #pragma unroll
            for (int v = 0; v < 8; ++v) {
                int key = k0 + t * 16 + hi * 8 + v;
                float sv = st[t][v] * 0.125f;            // 1/sqrt(64)
                if (key > rg) sv = -3.0e38f;             // causal mask
                s[t][v] = sv;
                mx = fmaxf(mx, sv);
            }
        mx = fmaxf(mx, __shfl_xor(mx, 16, 32));          // other 16 keys
        float mn = fmaxf(m, mx);
        float alpha = __expf(m - mn);
        float rs = 0.0f;
        #pragma unroll
        for (int t = 0; t < 2; ++t)
            #pragma unroll
            for (int v = 0; v < 8; ++v) {
                s[t][v] = __expf(s[t][v] - mn);
                rs += s[t][v];
            }
        rs += __shfl_xor(rs, 16, 32);
        l = l * alpha + rs;
        m = mn;
        #pragma unroll
        for (int c = 0; c < 4; ++c)
            #pragma unroll
            for (int v = 0; v < 8; ++v)
                acc[c][v] *= alpha;                       // uniform per lane

        // ---- P^T bounce: Ps[query][key], one uint4 store per tile ----
        #pragma unroll
        for (int t = 0; t < 2; ++t) {
            uint4 pk;
            uint32_t* pw = (uint32_t*)&pk;
            #pragma unroll
            for (int v = 0; v < 4; ++v)
                pw[v] = (uint32_t)f2bf_bits(s[t][2 * v]) |
                        ((uint32_t)f2bf_bits(s[t][2 * v + 1]) << 16);
            *(uint4*)(&Ps[wave][col][t * 16 + hi * 8]) = pk;
        }
        // B-fragment of P^T: b[e] = P[query=col][key = hi*8 + e (e<8), 16 + hi*8 + (e-8) (e>=8)]
        FragBF pb;
        {
            const unsigned short* pr = &Ps[wave][col][0];
            pb.q[0] = *(const uint4*)(pr + hi * 8);
            pb.q[1] = *(const uint4*)(pr + 16 + hi * 8);
        }
        // O^T += V^T P^T : A = V^T rows (row = dim = lane&15), K = 32 keys
        #pragma unroll
        for (int c = 0; c < 4; ++c) {
            FragBF a;
            const unsigned short* pr = &Vt[c * 16 + col][0];
            a.q[0] = *(const uint4*)(pr + hi * 8);
            a.q[1] = *(const uint4*)(pr + 16 + hi * 8);
            acc[c] = __builtin_amdgcn_wmma_f32_16x16x32_bf16(
                false, a.v, false, pb.v, (short)0, acc[c], false, false);
        }
        __syncthreads();
    }

    // ---- epilogue: O^T -> Og[query][dim], one uint4 store per d-chunk ----
    const float inv_l = 1.0f / l;
    unsigned short* orow = Og + (size_t)rg * D_MODEL + hoff;
    #pragma unroll
    for (int c = 0; c < 4; ++c) {
        uint4 pk;
        uint32_t* pw = (uint32_t*)&pk;
        #pragma unroll
        for (int v = 0; v < 4; ++v)
            pw[v] = (uint32_t)f2bf_bits(acc[c][2 * v] * inv_l) |
                    ((uint32_t)f2bf_bits(acc[c][2 * v + 1] * inv_l) << 16);
        *(uint4*)(orow + c * 16 + hi * 8) = pk;
    }
}

// ---------------------------------------------------------------------------
extern "C" void kernel_launch(void* const* d_in, const int* in_sizes, int n_in,
                              void* d_out, int out_size, void* d_ws, size_t ws_size,
                              hipStream_t stream)
{
    const float* x  = (const float*)d_in[0];
    const float* Wq = (const float*)d_in[1];
    const float* bq = (const float*)d_in[2];
    const float* Wk = (const float*)d_in[3];
    const float* bk = (const float*)d_in[4];
    const float* Wv = (const float*)d_in[5];
    const float* bv = (const float*)d_in[6];
    const float* Wo = (const float*)d_in[7];
    const float* bo = (const float*)d_in[8];

    const size_t tok = (size_t)SEQ * D_MODEL;
    unsigned short* Qb = (unsigned short*)d_ws;
    unsigned short* Kb = Qb + tok;
    unsigned short* Vb = Kb + tok;
    unsigned short* Ab = Vb + tok;           // total 32 MB of workspace

    dim3 gg(D_MODEL / 128, SEQ / 64);
    gemm_bias_kernel<false, true><<<gg, 256, 0, stream>>>(x, Wq, bq, Qb, SEQ, D_MODEL, D_MODEL);
    gemm_bias_kernel<false, true><<<gg, 256, 0, stream>>>(x, Wk, bk, Kb, SEQ, D_MODEL, D_MODEL);
    gemm_bias_kernel<false, true><<<gg, 256, 0, stream>>>(x, Wv, bv, Vb, SEQ, D_MODEL, D_MODEL);

    attn_kernel<<<dim3(SEQ / 64, NHEAD), 128, 0, stream>>>(Qb, Kb, Vb, Ab);

    gemm_bias_kernel<true, false><<<gg, 256, 0, stream>>>(Ab, Wo, bo, (float*)d_out,
                                                          SEQ, D_MODEL, D_MODEL);
}
